// RNNModel_59700045414406
// MI455X (gfx1250) — matrix operand order repaired and checked by reading, not compiled, not run on hardware
//
#include <hip/hip_runtime.h>

typedef _Float16 v8h  __attribute__((ext_vector_type(8)));
typedef _Float16 v16h __attribute__((ext_vector_type(16)));
typedef float    v8f  __attribute__((ext_vector_type(8)));

#define M_TOK 32768   // B*T
#define BATCH 16
#define TLEN  2048
#define EDIM  512
#define RDIM  1024
#define DDIM  128
#define LDS_STRIDE 40  // halves; 80B rows -> 16B-aligned row starts, conflict-light

// ---------------------------------------------------------------------------
// WMMA GEMM:  C[M,N] = A[M,K](f32) * Wt[N,K](f16, pre-transposed)
// (+bias)(+relu), f32 accum.
// Block tile 128x128, 256 thr = 8 waves, wave tile 32x64 = 2x4 wmma 16x16x32.
// Software-pipelined, double-buffered LDS:
//   frag loads -> global loads(next) -> 8 WMMAs -> LDS stores(next) -> barrier
// Requires: M%128==0, N%128==0, K%32==0 (true for all calls here).
// ---------------------------------------------------------------------------
template <int EPI>   // 0=none, 1=+bias, 2=+bias+relu
__global__ __launch_bounds__(256) void gemm_wmma_f16(
    const float* __restrict__ A, const _Float16* __restrict__ Wt,
    const float* __restrict__ bias, float* __restrict__ C,
    int M, int N, int K)
{
    __shared__ _Float16 As[2][128 * LDS_STRIDE];  // 128 rows x 32 K, f16
    __shared__ _Float16 Bs[2][128 * LDS_STRIDE];  // 128 cols x 32 K, f16 (K-minor)

    const int tid  = threadIdx.x;
    const int lane = tid & 31;
    const int wave = tid >> 5;
    const int wm   = wave & 3;   // 0..3 -> m offset 32*wm
    const int wn   = wave >> 2;  // 0..1 -> n offset 64*wn
    const int m0   = blockIdx.y * 128;
    const int n0   = blockIdx.x * 128;

    // staging index precompute
    const int ar[4] = { (tid + 0) >> 3, (tid + 256) >> 3, (tid + 512) >> 3, (tid + 768) >> 3 };
    const int ac    = (tid & 7) * 4;            // A: 4 x float4 per thread
    const int bcol  = tid >> 1;                 // B: one col, 16 halves per thread
    const int bseg  = (tid & 1) * 16;

    float4 areg[4];
    v8h    breg[2];

    auto gload = [&](int k0) {
#pragma unroll
        for (int it = 0; it < 4; ++it)
            areg[it] = *(const float4*)(A + (size_t)(m0 + ar[it]) * K + k0 + ac);
        const _Float16* bp = Wt + (size_t)(n0 + bcol) * K + k0 + bseg;
        breg[0] = *(const v8h*)bp;
        breg[1] = *(const v8h*)(bp + 8);
    };
    auto sstore = [&](int buf) {
#pragma unroll
        for (int it = 0; it < 4; ++it) {
            _Float16* dst = &As[buf][ar[it] * LDS_STRIDE + ac];
            dst[0] = (_Float16)areg[it].x; dst[1] = (_Float16)areg[it].y;
            dst[2] = (_Float16)areg[it].z; dst[3] = (_Float16)areg[it].w;
        }
        _Float16* dst = &Bs[buf][bcol * LDS_STRIDE + bseg];
        *(v8h*)dst       = breg[0];
        *(v8h*)(dst + 8) = breg[1];
    };

    v8f acc[2][4] = {};
    const int rsel = lane & 15;
    const int kba  = (lane < 16) ? 0 : 8;   // A frag: K {0..7,16..23} | {8..15,24..31}
    const int kbb  = (lane < 16) ? 0 : 8;   // B frag: K {0..7,16..23} | {8..15,24..31}

    gload(0);
    sstore(0);
    __syncthreads();

    const int nk = K >> 5;
    for (int s = 0; s < nk; ++s) {
        const int buf = s & 1;
        // 1) gather fragments for this K-step into registers
        v16h afrag[2], bfrag[4];
#pragma unroll
        for (int i = 0; i < 2; ++i) {
            const _Float16* p = &As[buf][(wm * 32 + i * 16 + rsel) * LDS_STRIDE + kba];
            v8h lo = *(const v8h*)p;
            v8h hi = *(const v8h*)(p + 16);
            afrag[i] = __builtin_shufflevector(lo, hi, 0,1,2,3,4,5,6,7,8,9,10,11,12,13,14,15);
        }
#pragma unroll
        for (int j = 0; j < 4; ++j) {
            const _Float16* p = &Bs[buf][(wn * 64 + j * 16 + rsel) * LDS_STRIDE + kbb];
            v8h lo = *(const v8h*)p;
            v8h hi = *(const v8h*)(p + 16);
            bfrag[j] = __builtin_shufflevector(lo, hi, 0,1,2,3,4,5,6,7,8,9,10,11,12,13,14,15);
        }
        // 2) issue next K-step's global loads (batched; latency hidden by WMMAs)
        if (s + 1 < nk) gload((s + 1) * 32);
        // 3) 8 WMMAs on registers
#pragma unroll
        for (int i = 0; i < 2; ++i)
#pragma unroll
            for (int j = 0; j < 4; ++j)
                acc[i][j] = __builtin_amdgcn_wmma_f32_16x16x32_f16(
                    false, afrag[i], false, bfrag[j], (short)0, acc[i][j], false, false);
        // 4) drain global loads into the other LDS buffer
        if (s + 1 < nk) sstore(buf ^ 1);
        __syncthreads();
    }

    // Epilogue. C/D layout: VGPR e -> row e (lanes 0-15) / e+8 (lanes 16-31)
    const int roff = (lane < 16) ? 0 : 8;
#pragma unroll
    for (int i = 0; i < 2; ++i)
#pragma unroll
        for (int j = 0; j < 4; ++j) {
            const int col = n0 + wn * 64 + j * 16 + rsel;
            const int row = m0 + wm * 32 + i * 16 + roff;
            size_t base = (size_t)row * N + col;
            float bv = 0.0f;
            if (EPI >= 1) bv = bias[col];
#pragma unroll
            for (int e = 0; e < 8; ++e) {
                float v = acc[i][j][e];
                if (EPI >= 1) v += bv;
                if (EPI == 2) v = fmaxf(v, 0.0f);
                C[base + (size_t)e * N] = v;
            }
        }
}

// ---------------------------------------------------------------------------
// One-time prepass: W[K,N] fp32 -> Wt[N,K] f16 (transposed)
// ---------------------------------------------------------------------------
__global__ __launch_bounds__(256) void cvt_t_f16_kernel(const float* __restrict__ src,
                                                        _Float16* __restrict__ dst,
                                                        int K, int N)
{
    int i = blockIdx.x * blockDim.x + threadIdx.x;   // over N*K (output-major)
    if (i < N * K) {
        int n = i / K;
        int k = i - n * K;
        dst[i] = (_Float16)src[(size_t)k * N + n];
    }
}

// ---------------------------------------------------------------------------
// Prep: bias_feat = b_feat + sys_emb[np]; h0 = relu(se @ W_h0 + b_h0)
// ---------------------------------------------------------------------------
__global__ void prep_kernel(const float* __restrict__ b_feat,
                            const float* __restrict__ sys_emb,
                            const float* __restrict__ W_h0,
                            const float* __restrict__ b_h0,
                            const int* __restrict__ np_p,
                            float* __restrict__ bias_feat,
                            float* __restrict__ h0)
{
    int i = blockIdx.x * blockDim.x + threadIdx.x;
    int np = np_p[0];
    const float* se = sys_emb + (size_t)np * EDIM;
    if (i < EDIM) bias_feat[i] = b_feat[i] + se[i];
    if (i < RDIM) {
        float s = b_h0[i];
        for (int e = 0; e < EDIM; ++e) s += se[e] * W_h0[(size_t)e * RDIM + i];
        h0[i] = fmaxf(s, 0.0f);
    }
}

// ---------------------------------------------------------------------------
// SRU recurrence: one thread per (b, channel); sequential over t.
// U is (B*T, k*R) row-major; channel reads are coalesced across the wave.
// ---------------------------------------------------------------------------
__global__ __launch_bounds__(256) void sru_scan(
    const float* __restrict__ U, const float* __restrict__ xin,
    const float* __restrict__ b_f, const float* __restrict__ b_r,
    const float* __restrict__ h0, float* __restrict__ h, int k)
{
    const int idx = blockIdx.x * blockDim.x + threadIdx.x;  // 0..16383
    const int b = idx >> 10;
    const int j = idx & (RDIM - 1);
    float c = h0[j];
    const float bf = b_f[j], br = b_r[j];
    const size_t rs = (size_t)k * RDIM;
    const float* Ub = U + (size_t)b * TLEN * rs;
    const float* xb = xin ? (xin + (size_t)b * TLEN * RDIM) : nullptr;
    float* hb = h + (size_t)b * TLEN * RDIM;
    const float RESCALE = 1.41421356237f;
    for (int t = 0; t < TLEN; ++t) {
        const float* u = Ub + (size_t)t * rs;
        float z = u[j];
        float f = 1.0f / (1.0f + __expf(-(u[RDIM + j] + bf)));
        float r = 1.0f / (1.0f + __expf(-(u[2 * RDIM + j] + br)));
        float res = (k == 4) ? u[3 * RDIM + j] : xb[(size_t)t * RDIM + j];
        c = f * c + (1.0f - f) * z;
        float g = fmaxf(c, 0.0f);
        hb[(size_t)t * RDIM + j] = r * g + (1.0f - r) * res * RESCALE;
    }
}

// ---------------------------------------------------------------------------
// Deterministic 2-pass MSE loss
// ---------------------------------------------------------------------------
__global__ __launch_bounds__(256) void loss_partial(const float* __restrict__ y,
                                                    const float* __restrict__ yp,
                                                    float* __restrict__ partials,
                                                    size_t n)
{
    __shared__ float sdata[256];
    float s = 0.0f;
    for (size_t i = (size_t)blockIdx.x * 256 + threadIdx.x; i < n; i += (size_t)256 * gridDim.x) {
        float d = y[i] - yp[i];
        s += d * d;
    }
    sdata[threadIdx.x] = s;
    __syncthreads();
    for (int off = 128; off > 0; off >>= 1) {
        if (threadIdx.x < off) sdata[threadIdx.x] += sdata[threadIdx.x + off];
        __syncthreads();
    }
    if (threadIdx.x == 0) partials[blockIdx.x] = sdata[0];
}

__global__ void loss_final(const float* __restrict__ partials, float* __restrict__ out, float inv_n)
{
    __shared__ float sdata[256];
    sdata[threadIdx.x] = partials[threadIdx.x];
    __syncthreads();
    for (int off = 128; off > 0; off >>= 1) {
        if (threadIdx.x < off) sdata[threadIdx.x] += sdata[threadIdx.x + off];
        __syncthreads();
    }
    if (threadIdx.x == 0) out[0] = sdata[0] * inv_n;
}

// ---------------------------------------------------------------------------
static void launch_gemm(int epi, const float* A, const _Float16* Wt, const float* bias,
                        float* C, int M, int N, int K, hipStream_t s)
{
    dim3 g(N / 128, M / 128), b(256);
    if (epi == 0)      gemm_wmma_f16<0><<<g, b, 0, s>>>(A, Wt, nullptr, C, M, N, K);
    else if (epi == 1) gemm_wmma_f16<1><<<g, b, 0, s>>>(A, Wt, bias, C, M, N, K);
    else               gemm_wmma_f16<2><<<g, b, 0, s>>>(A, Wt, bias, C, M, N, K);
}

static void launch_cvt_t(const float* src, _Float16* dst, int K, int N, hipStream_t s)
{
    int n = K * N;
    cvt_t_f16_kernel<<<(n + 255) / 256, 256, 0, s>>>(src, dst, K, N);
}

extern "C" void kernel_launch(void* const* d_in, const int* in_sizes, int n_in,
                              void* d_out, int out_size, void* d_ws, size_t ws_size,
                              hipStream_t stream)
{
    // Input order (setup_inputs dict, params flattened in insertion order):
    const float* x       = (const float*)d_in[0];   // (16,2048,128)
    const float* y       = (const float*)d_in[1];   // (16,2048,128)
    const float* W_feat  = (const float*)d_in[2];   // (128,512)
    const float* b_feat  = (const float*)d_in[3];   // (512)
    const float* sys_emb = (const float*)d_in[4];   // (7,512)
    const float* W_pre   = (const float*)d_in[5];   // (512,512)
    const float* b_pre   = (const float*)d_in[6];   // (512)
    const float* W_h0    = (const float*)d_in[7];   // (512,1024)
    const float* b_h0    = (const float*)d_in[8];   // (1024)
    const float* sruW[4] = {(const float*)d_in[9],  (const float*)d_in[12],
                            (const float*)d_in[15], (const float*)d_in[18]};
    const float* sruBf[4]= {(const float*)d_in[10], (const float*)d_in[13],
                            (const float*)d_in[16], (const float*)d_in[19]};
    const float* sruBr[4]= {(const float*)d_in[11], (const float*)d_in[14],
                            (const float*)d_in[17], (const float*)d_in[20]};
    const float* W_post  = (const float*)d_in[21];  // (1024,512)
    const float* b_post  = (const float*)d_in[22];  // (512)
    const float* W_rest  = (const float*)d_in[23];  // (512,128)
    const float* b_rest  = (const float*)d_in[24];  // (128)
    const int*   np_p    = (const int*)d_in[25];    // scalar 6

    float* out_f = (float*)d_out;
    float* yhat  = out_f + 1;                        // y_ goes after loss scalar

    // Workspace layout
    float* ws        = (float*)d_ws;
    float* bufA      = ws;                                   // M x 1024
    float* bufB      = bufA + (size_t)M_TOK * RDIM;          // M x 1024
    float* bufU      = bufB + (size_t)M_TOK * RDIM;          // M x 4096
    float* bias_feat = bufU + (size_t)M_TOK * 4 * RDIM;      // 512
    float* h0v       = bias_feat + EDIM;                     // 1024
    float* partials  = h0v + RDIM;                           // 256

    _Float16* w16 = (_Float16*)(partials + 256);             // transposed f16 weights
    _Float16* w16_feat = w16;                                //  512x128   (N x K)
    _Float16* w16_pre  = w16_feat + (size_t)DDIM * EDIM;     //  512x512
    _Float16* w16_sru0 = w16_pre  + (size_t)EDIM * EDIM;     // 4096x512
    _Float16* w16_sru1 = w16_sru0 + (size_t)EDIM * 4 * RDIM; // 3072x1024
    _Float16* w16_sru2 = w16_sru1 + (size_t)RDIM * 3 * RDIM;
    _Float16* w16_sru3 = w16_sru2 + (size_t)RDIM * 3 * RDIM;
    _Float16* w16_post = w16_sru3 + (size_t)RDIM * 3 * RDIM; //  512x1024
    _Float16* w16_rest = w16_post + (size_t)RDIM * EDIM;     //  128x512
    _Float16* w16_sru[4] = { w16_sru0, w16_sru1, w16_sru2, w16_sru3 };

    // 0) prepasses: small vectors + transposed f16 weight conversion
    prep_kernel<<<4, 256, 0, stream>>>(b_feat, sys_emb, W_h0, b_h0, np_p, bias_feat, h0v);
    launch_cvt_t(W_feat,  w16_feat, DDIM, EDIM, stream);
    launch_cvt_t(W_pre,   w16_pre,  EDIM, EDIM, stream);
    launch_cvt_t(sruW[0], w16_sru0, EDIM, 4 * RDIM, stream);
    launch_cvt_t(sruW[1], w16_sru1, RDIM, 3 * RDIM, stream);
    launch_cvt_t(sruW[2], w16_sru2, RDIM, 3 * RDIM, stream);
    launch_cvt_t(sruW[3], w16_sru3, RDIM, 3 * RDIM, stream);
    launch_cvt_t(W_post,  w16_post, RDIM, EDIM, stream);
    launch_cvt_t(W_rest,  w16_rest, EDIM, DDIM, stream);

    // 1) feat: x(M,128) @ W_feat -> bufA(M,512)  (+ b_feat + sys_emb)
    launch_gemm(1, x, w16_feat, bias_feat, bufA, M_TOK, EDIM, DDIM, stream);

    // 2) pre: relu(bufA @ W_pre + b_pre) -> bufB(M,512)
    launch_gemm(2, bufA, w16_pre, b_pre, bufB, M_TOK, EDIM, EDIM, stream);

    // 3) SRU layer 0: U = bufB @ W0 (512 -> 4096); scan -> bufA(M,1024)
    launch_gemm(0, bufB, w16_sru0, nullptr, bufU, M_TOK, 4 * RDIM, EDIM, stream);
    sru_scan<<<64, 256, 0, stream>>>(bufU, nullptr, sruBf[0], sruBr[0], h0v, bufA, 4);

    // 4) SRU layers 1..3 (1024 -> 3072), residual = layer input
    float* cur = bufA;
    float* nxt = bufB;
    for (int l = 1; l < 4; ++l) {
        launch_gemm(0, cur, w16_sru[l], nullptr, bufU, M_TOK, 3 * RDIM, RDIM, stream);
        sru_scan<<<64, 256, 0, stream>>>(bufU, cur, sruBf[l], sruBr[l], h0v, nxt, 3);
        float* t = cur; cur = nxt; nxt = t;
    }
    // after loop: cur = bufB holds rnn_out(M,1024)

    // 5) post: relu(cur @ W_post + b_post) -> nxt(M,512)
    launch_gemm(2, cur, w16_post, b_post, nxt, M_TOK, 512, RDIM, stream);

    // 6) rest: nxt @ W_rest + b_rest -> yhat(M,128)
    launch_gemm(1, nxt, w16_rest, b_rest, yhat, M_TOK, DDIM, 512, stream);

    // 7) deterministic MSE loss -> out[0]
    const size_t nelem = (size_t)M_TOK * DDIM;
    loss_partial<<<256, 256, 0, stream>>>(y, yhat, partials, nelem);
    loss_final<<<1, 256, 0, stream>>>(partials, out_f, 1.0f / (float)nelem);
}
